// DiTBlock_15015205667478
// MI455X (gfx1250) — matrix operand order repaired and checked by reading, not compiled, not run on hardware
//
#include <hip/hip_runtime.h>
#include <hip/hip_bf16.h>
#include <math.h>

// ---------------------------------------------------------------------------
// Types / helpers
// ---------------------------------------------------------------------------
typedef __attribute__((ext_vector_type(16))) __bf16 v16bf;
typedef __attribute__((ext_vector_type(8)))  float  v8f;

struct U256 { uint4 lo, hi; };   // 32-byte carrier for a v16bf fragment

#define LDSB 40   // LDS row stride in bf16 elems (80B: 16B-aligned, spreads banks)

__device__ __forceinline__ unsigned short f32_to_bf16(float f) {
    unsigned u = __builtin_bit_cast(unsigned, f);
    u += 0x7FFFu + ((u >> 16) & 1u);          // round-to-nearest-even
    return (unsigned short)(u >> 16);
}

// CDNA5 async global->LDS copy (ASYNCcnt), GV addressing: 16B per lane.
// ldsOff = byte offset within workgroup LDS (flat LDS addr low 32 bits).
__device__ __forceinline__ void async_b128(unsigned ldsOff, const void* g) {
    asm volatile("global_load_async_to_lds_b128 %0, %1, off"
                 :: "v"(ldsOff), "v"(g)
                 : "memory");
}

__device__ __forceinline__ unsigned lds_off(const void* p) {
    return (unsigned)(size_t)p;   // flat LDS aperture: addr[31:0] is the LDS offset
}

// ---------------------------------------------------------------------------
// Weight transpose + fp32 -> bf16 convert.  W is K x N row-major (fp32),
// Wt is N x K row-major (bf16).
// ---------------------------------------------------------------------------
__global__ __launch_bounds__(256) void transpose_bf16_kernel(
    const float* __restrict__ W, unsigned short* __restrict__ Wt, int K, int N) {
    __shared__ float tile[32][33];
    const int n = blockIdx.x * 32 + threadIdx.x;
    #pragma unroll
    for (int i = 0; i < 4; ++i) {
        const int k = blockIdx.y * 32 + threadIdx.y + i * 8;
        tile[threadIdx.y + i * 8][threadIdx.x] = W[(size_t)k * N + n];
    }
    __syncthreads();
    const int ko = blockIdx.y * 32 + threadIdx.x;
    #pragma unroll
    for (int i = 0; i < 4; ++i) {
        const int no = blockIdx.x * 32 + threadIdx.y + i * 8;
        Wt[(size_t)no * K + ko] = f32_to_bf16(tile[threadIdx.x][threadIdx.y + i * 8]);
    }
}

// ---------------------------------------------------------------------------
// cond modulation: out[m][b][d] = cond[b] @ w_m + b_m  (tanh*0.1 for m=2,5)
// grid = 6 mats * 4 batch * 4 chunks = 96 blocks, 256 threads
// ---------------------------------------------------------------------------
struct CondArgs {
    const float* cond;
    const float* wt[6];
    const float* bs[6];
    float* out;
};

__global__ __launch_bounds__(256) void cond_kernel(CondArgs a) {
    const int id = blockIdx.x;
    const int m = id >> 4;
    const int b = (id >> 2) & 3;
    const int chunk = id & 3;
    __shared__ float cs[1024];
    #pragma unroll
    for (int i = 0; i < 4; ++i)
        cs[threadIdx.x + i * 256] = a.cond[(b << 10) + threadIdx.x + i * 256];
    __syncthreads();
    const int col = (chunk << 8) + threadIdx.x;
    const float* w = a.wt[m];
    float acc = a.bs[m][col];
    for (int k = 0; k < 1024; ++k)
        acc = fmaf(cs[k], w[(k << 10) + col], acc);
    if (m == 2 || m == 5) acc = tanhf(acc) * 0.1f;
    a.out[(size_t)((m << 2) + b) * 1024 + col] = acc;
}

// ---------------------------------------------------------------------------
// LayerNorm + adaLN modulation -> bf16.  One block per token row (D=1024).
// y = LN(x)*(1+gamma[b]) + beta[b]
// ---------------------------------------------------------------------------
__global__ __launch_bounds__(256) void ln_mod_kernel(
    const float* __restrict__ x, const float* __restrict__ lng,
    const float* __restrict__ lnb, const float* __restrict__ gamma,
    const float* __restrict__ beta, unsigned short* __restrict__ y) {
    const int row = blockIdx.x;
    const int t = threadIdx.x;
    const int b = row >> 10;
    const float4 v = ((const float4*)(x + (size_t)row * 1024))[t];
    __shared__ float red[256];
    red[t] = v.x + v.y + v.z + v.w;
    __syncthreads();
    for (int s = 128; s > 0; s >>= 1) { if (t < s) red[t] += red[t + s]; __syncthreads(); }
    const float mu = red[0] * (1.0f / 1024.0f);
    __syncthreads();
    const float d0 = v.x - mu, d1 = v.y - mu, d2 = v.z - mu, d3 = v.w - mu;
    red[t] = d0 * d0 + d1 * d1 + d2 * d2 + d3 * d3;
    __syncthreads();
    for (int s = 128; s > 0; s >>= 1) { if (t < s) red[t] += red[t + s]; __syncthreads(); }
    const float rstd = rsqrtf(red[0] * (1.0f / 1024.0f) + 1e-5f);
    const int c0 = t * 4;
    const float* gm = gamma + (b << 10);
    const float* bt = beta + (b << 10);
    const float dd[4] = {d0, d1, d2, d3};
    unsigned short o[4];
    #pragma unroll
    for (int j = 0; j < 4; ++j) {
        const int c = c0 + j;
        float val = dd[j] * rstd * lng[c] + lnb[c];
        val = val * (1.0f + gm[c]) + bt[c];
        o[j] = f32_to_bf16(val);
    }
    uint2 pk;
    pk.x = (unsigned)o[0] | ((unsigned)o[1] << 16);
    pk.y = (unsigned)o[2] | ((unsigned)o[3] << 16);
    ((uint2*)(y + (size_t)row * 1024))[t] = pk;
}

// ---------------------------------------------------------------------------
// In-place softmax over a 1024-wide row of f32 scores; writes bf16 probs into
// the first half of the same row (probs row stride = 2048 bf16 elements).
// ---------------------------------------------------------------------------
__global__ __launch_bounds__(256) void softmax_kernel(float* __restrict__ scores) {
    const int row = blockIdx.x;
    const int t = threadIdx.x;
    float* base = scores + (size_t)row * 1024;
    const float4 v = ((const float4*)base)[t];
    __shared__ float red[256];
    red[t] = fmaxf(fmaxf(v.x, v.y), fmaxf(v.z, v.w));
    __syncthreads();
    for (int s = 128; s > 0; s >>= 1) { if (t < s) red[t] = fmaxf(red[t], red[t + s]); __syncthreads(); }
    const float mx = red[0];
    __syncthreads();
    const float e0 = __expf(v.x - mx), e1 = __expf(v.y - mx);
    const float e2 = __expf(v.z - mx), e3 = __expf(v.w - mx);
    red[t] = e0 + e1 + e2 + e3;
    __syncthreads();
    for (int s = 128; s > 0; s >>= 1) { if (t < s) red[t] += red[t + s]; __syncthreads(); }
    const float inv = 1.0f / red[0];
    __syncthreads();   // all reads of this row complete before in-place writes
    uint2 pk;
    pk.x = (unsigned)f32_to_bf16(e0 * inv) | ((unsigned)f32_to_bf16(e1 * inv) << 16);
    pk.y = (unsigned)f32_to_bf16(e2 * inv) | ((unsigned)f32_to_bf16(e3 * inv) << 16);
    ((uint2*)base)[t] = pk;
}

// ---------------------------------------------------------------------------
// Generic bf16 WMMA GEMM with async double-buffered LDS pipeline:
//   C(MxN) = A(MxK) * B(KxN) [+ epilogue]
//   TRANSB=1: B given as Bt, N x K row-major; A and B tiles filled with
//             global_load_async_to_lds_b128 (ASYNCcnt), double buffered.
//   TRANSB=0: B given as K x N row-major, gathered via VGPR into a
//             transposed LDS tile; A still uses the async path.
// 256 threads = 8 waves (2x4); each wave computes 64 x (BN/4) with
// v_wmma_f32_16x16x32_bf16. grid.z = head batch; strides in elements.
// EPI: 0 = +bias -> bf16 | 1 = *scale -> f32 | 2 = -> bf16
//      3 = +bias, exact GELU -> bf16 | 4 = resid + (acc+bias)*alpha[b] -> f32
// ---------------------------------------------------------------------------
struct GemmP {
    const unsigned short* A;
    const unsigned short* B;
    const float* bias;
    const float* resid;
    const float* alpha;
    float* outF;
    unsigned short* outB;
    int K, lda, ldb, ldc;
    size_t strideAz, strideBz, strideCz;
    float scale;
};

template <int BM, int BN, bool TRANSB, int EPI>
__global__ __launch_bounds__(256) void gemm_bf16_wmma(GemmP p) {
    static_assert(BM == 128, "BM fixed at 128");
    constexpr int WN = BN / 4;    // wave N extent
    constexpr int NI = WN / 16;   // N sub-tiles per wave
    constexpr int MI = 4;         // 64 / 16
    constexpr int SEGB = (BN * 32) / 256;             // u16 per thread for B tile
    constexpr int APT = TRANSB ? (2 + SEGB / 8) : 2;  // async ops per tile per wave

    __shared__ __align__(16) unsigned short As[2][BM * LDSB];
    __shared__ __align__(16) unsigned short Bs[2][BN * LDSB];

    const int t = threadIdx.x;
    const int lane = t & 31;
    const int waveM = (t >> 5) >> 2;   // 0..1
    const int waveN = (t >> 5) & 3;    // 0..3
    const int lrow = lane & 15;
    const int kh = lane >> 4;

    const int m0 = blockIdx.y * BM;
    const int n0 = blockIdx.x * BN;
    const int z = blockIdx.z;

    const unsigned short* Ab = p.A + (size_t)z * p.strideAz;
    const unsigned short* Bb = p.B + (size_t)z * p.strideBz;

    // per-thread stage addressing (A: 32B per thread, two 16B chunks 16 elems apart in K)
    const int ar = t >> 1, asg = t & 1;
    const unsigned short* aSrc = Ab + (size_t)(m0 + ar) * p.lda + asg * 16;
    const unsigned aDst0 = lds_off(&As[0][ar * LDSB + asg * 16]);
    const unsigned aDst1 = lds_off(&As[1][ar * LDSB + asg * 16]);

    // B stage addressing (TRANSB=1 path)
    constexpr int SPR = 32 / SEGB;
    const int br = t / SPR, bsg = t % SPR;
    const unsigned short* bSrc = Bb + (size_t)(n0 + br) * p.ldb + bsg * SEGB;
    const unsigned bDst0 = lds_off(&Bs[0][br * LDSB + bsg * SEGB]);
    const unsigned bDst1 = lds_off(&Bs[1][br * LDSB + bsg * SEGB]);

    auto stage = [&](int buf, int k0) {
        const unsigned ad = buf ? aDst1 : aDst0;
        async_b128(ad, aSrc + k0);
        async_b128(ad + 32, aSrc + k0 + 16);
        if constexpr (TRANSB) {
            const unsigned bd = buf ? bDst1 : bDst0;
            async_b128(bd, bSrc + k0);
            if constexpr (SEGB == 16) async_b128(bd + 32, bSrc + k0 + 16);
        } else {
            // gather K x BN slab into transposed LDS tile Bs[n][k]
            const int kk = t >> 3;
            const int nr = (t & 7) * SEGB;
            union { uint4 u[2]; unsigned short s[16]; } tmp;
            const uint4* src = (const uint4*)(Bb + (size_t)(k0 + kk) * p.ldb + n0 + nr);
            tmp.u[0] = src[0];
            if constexpr (SEGB == 16) tmp.u[1] = src[1];
            #pragma unroll
            for (int i = 0; i < SEGB; ++i)
                Bs[buf][(nr + i) * LDSB + kk] = tmp.s[i];
        }
    };

    v8f acc[MI][NI] = {};
    const int iters = p.K >> 5;

    stage(0, 0);
    for (int i = 0; i < iters; ++i) {
        const int nxt = i + 1;
        if (nxt < iters) {
            stage(nxt & 1, nxt * 32);
            asm volatile("s_wait_asynccnt %0" :: "n"(APT));  // tile i resident
        } else {
            asm volatile("s_wait_asynccnt %0" :: "n"(0));
        }
        __syncthreads();

        const int buf = i & 1;
        // ---- fragments (ISA 7.12.2 layouts)
        v16bf afr[MI];
        #pragma unroll
        for (int mi = 0; mi < MI; ++mi) {
            // A 16x32: lane-half kh holds K = kh*8+e (e<8) and 16+kh*8+(e-8)
            const unsigned short* pa = &As[buf][(waveM * 64 + mi * 16 + lrow) * LDSB + kh * 8];
            U256 r;
            r.lo = *(const uint4*)pa;
            r.hi = *(const uint4*)(pa + 16);
            afr[mi] = __builtin_bit_cast(v16bf, r);
        }
        v16bf bfr[NI];
        #pragma unroll
        for (int ni = 0; ni < NI; ++ni) {
            // B 32x16: lane-half kh holds K = kh*8+e (e<8) and 16+kh*8+(e-8)
            const unsigned short* pb = &Bs[buf][(waveN * WN + ni * 16 + lrow) * LDSB + kh * 8];
            U256 r;
            r.lo = *(const uint4*)pb;
            r.hi = *(const uint4*)(pb + 16);
            bfr[ni] = __builtin_bit_cast(v16bf, r);
        }
        #pragma unroll
        for (int mi = 0; mi < MI; ++mi)
            #pragma unroll
            for (int ni = 0; ni < NI; ++ni)
                acc[mi][ni] = __builtin_amdgcn_wmma_f32_16x16x32_bf16(
                    false, afr[mi], false, bfr[ni], (short)0, acc[mi][ni], false, false);
        __syncthreads();
    }

    // ---- epilogue (C/D: VGPR r, lanes 0-15 -> M=r, lanes 16-31 -> M=8+r)
    const size_t zc = (size_t)z * p.strideCz;
    #pragma unroll
    for (int mi = 0; mi < MI; ++mi) {
        #pragma unroll
        for (int ni = 0; ni < NI; ++ni) {
            #pragma unroll
            for (int r = 0; r < 8; ++r) {
                const int rowg = m0 + waveM * 64 + mi * 16 + kh * 8 + r;
                const int colg = n0 + waveN * WN + ni * 16 + lrow;
                float v = acc[mi][ni][r];
                const size_t idx = zc + (size_t)rowg * p.ldc + colg;
                if constexpr (EPI == 0) {
                    p.outB[idx] = f32_to_bf16(v + p.bias[colg]);
                } else if constexpr (EPI == 1) {
                    p.outF[idx] = v * p.scale;
                } else if constexpr (EPI == 2) {
                    p.outB[idx] = f32_to_bf16(v);
                } else if constexpr (EPI == 3) {
                    v += p.bias[colg];
                    const float g = 0.5f * v * (1.0f + erff(v * 0.70710678118654752f));
                    p.outB[idx] = f32_to_bf16(g);
                } else {  // EPI == 4
                    v += p.bias[colg];
                    const int bb = rowg >> 10;
                    p.outF[idx] = p.resid[idx] + v * p.alpha[(bb << 10) + colg];
                }
            }
        }
    }
}

// ---------------------------------------------------------------------------
// Host orchestration
// ---------------------------------------------------------------------------
extern "C" void kernel_launch(void* const* d_in, const int* in_sizes, int n_in,
                              void* d_out, int out_size, void* d_ws, size_t ws_size,
                              hipStream_t stream) {
    (void)in_sizes; (void)n_in; (void)out_size; (void)ws_size;
    const float* x      = (const float*)d_in[0];
    const float* cond   = (const float*)d_in[1];
    const float* g1_w   = (const float*)d_in[2];  const float* g1_b = (const float*)d_in[3];
    const float* b1_w   = (const float*)d_in[4];  const float* b1_b = (const float*)d_in[5];
    const float* a1_w   = (const float*)d_in[6];  const float* a1_b = (const float*)d_in[7];
    const float* g2_w   = (const float*)d_in[8];  const float* g2_b = (const float*)d_in[9];
    const float* b2_w   = (const float*)d_in[10]; const float* b2_b = (const float*)d_in[11];
    const float* a2_w   = (const float*)d_in[12]; const float* a2_b = (const float*)d_in[13];
    const float* ln1_g  = (const float*)d_in[14]; const float* ln1_b = (const float*)d_in[15];
    const float* ln2_g  = (const float*)d_in[16]; const float* ln2_b = (const float*)d_in[17];
    const float* qkv_w  = (const float*)d_in[18]; const float* qkv_b = (const float*)d_in[19];
    const float* proj_w = (const float*)d_in[20]; const float* proj_b = (const float*)d_in[21];
    const float* fc1_w  = (const float*)d_in[22]; const float* fc1_b = (const float*)d_in[23];
    const float* fc2_w  = (const float*)d_in[24]; const float* fc2_b = (const float*)d_in[25];
    float* out = (float*)d_out;

    // workspace carve-up (~160 MiB, all sizes 256B multiples)
    char* ws = (char*)d_ws;
    size_t off = 0;
    auto take = [&](size_t bytes) { char* p = ws + off; off += (bytes + 255) & ~(size_t)255; return p; };
    unsigned short* Wqkvt  = (unsigned short*)take(3072ull * 1024 * 2);
    unsigned short* Wprojt = (unsigned short*)take(1024ull * 1024 * 2);
    unsigned short* Wfc1t  = (unsigned short*)take(4096ull * 1024 * 2);
    unsigned short* Wfc2t  = (unsigned short*)take(1024ull * 4096 * 2);
    unsigned short* ybf    = (unsigned short*)take(4096ull * 1024 * 2);
    unsigned short* qkvb   = (unsigned short*)take(4096ull * 3072 * 2);
    unsigned short* attnb  = (unsigned short*)take(4096ull * 1024 * 2);
    unsigned short* h1     = (unsigned short*)take(4096ull * 4096 * 2);
    float*          scores = (float*)take(16ull * 1024 * 1024 * 4);
    float*          mod    = (float*)take(6ull * 4 * 1024 * 4);

    const dim3 tb(32, 8, 1);
    transpose_bf16_kernel<<<dim3(3072 / 32, 1024 / 32), tb, 0, stream>>>(qkv_w,  Wqkvt,  1024, 3072);
    transpose_bf16_kernel<<<dim3(1024 / 32, 1024 / 32), tb, 0, stream>>>(proj_w, Wprojt, 1024, 1024);
    transpose_bf16_kernel<<<dim3(4096 / 32, 1024 / 32), tb, 0, stream>>>(fc1_w,  Wfc1t,  1024, 4096);
    transpose_bf16_kernel<<<dim3(1024 / 32, 4096 / 32), tb, 0, stream>>>(fc2_w,  Wfc2t,  4096, 1024);

    CondArgs ca;
    ca.cond = cond; ca.out = mod;
    ca.wt[0] = g1_w; ca.bs[0] = g1_b;
    ca.wt[1] = b1_w; ca.bs[1] = b1_b;
    ca.wt[2] = a1_w; ca.bs[2] = a1_b;   // tanh * 0.1
    ca.wt[3] = g2_w; ca.bs[3] = g2_b;
    ca.wt[4] = b2_w; ca.bs[4] = b2_b;
    ca.wt[5] = a2_w; ca.bs[5] = a2_b;   // tanh * 0.1
    cond_kernel<<<96, 256, 0, stream>>>(ca);

    // y = LN1(x) * (1+gamma1) + beta1  -> bf16
    ln_mod_kernel<<<4096, 256, 0, stream>>>(x, ln1_g, ln1_b, mod, mod + 4096, ybf);

    // QKV: [4096,1024] x [1024,3072] (+bias) -> bf16 [B,N,3,H,dh]
    {
        GemmP g{};
        g.A = ybf; g.B = Wqkvt; g.bias = qkv_b; g.outB = qkvb;
        g.K = 1024; g.lda = 1024; g.ldb = 1024; g.ldc = 3072; g.scale = 1.0f;
        gemm_bf16_wmma<128, 128, true, 0><<<dim3(3072 / 128, 4096 / 128, 1), 256, 0, stream>>>(g);
    }

    // attention, one batch at a time (reuses the score buffer)
    for (int b = 0; b < 4; ++b) {
        const size_t qb = (size_t)b * 1024 * 3072;
        {   // scores[h] = (Q_h @ K_h^T) * dh^-0.5  (K slab is already "Bt")
            GemmP s{};
            s.A = qkvb + qb;            s.lda = 3072; s.strideAz = 64;
            s.B = qkvb + qb + 1024;     s.ldb = 3072; s.strideBz = 64;
            s.outF = scores;            s.ldc = 1024; s.strideCz = 1024ull * 1024;
            s.K = 64; s.scale = 0.125f;
            gemm_bf16_wmma<128, 128, true, 1><<<dim3(8, 8, 16), 256, 0, stream>>>(s);
        }
        softmax_kernel<<<16 * 1024, 256, 0, stream>>>(scores);
        {   // attn_out[h] = P_h @ V_h  (V is K x N row-major -> TRANSB=0)
            GemmP a{};
            a.A = (const unsigned short*)scores; a.lda = 2048; a.strideAz = 2048ull * 1024;
            a.B = qkvb + qb + 2048;              a.ldb = 3072; a.strideBz = 64;
            a.outB = attnb + (size_t)b * 1024 * 1024; a.ldc = 1024; a.strideCz = 64;
            a.K = 1024; a.scale = 1.0f;
            gemm_bf16_wmma<128, 64, false, 2><<<dim3(1, 8, 16), 256, 0, stream>>>(a);
        }
    }

    // x1 = x + (attn @ proj_w + proj_b) * alpha1  -> d_out (f32)
    {
        GemmP g{};
        g.A = attnb; g.B = Wprojt; g.bias = proj_b; g.resid = x; g.alpha = mod + 2 * 4096;
        g.outF = out; g.K = 1024; g.lda = 1024; g.ldb = 1024; g.ldc = 1024;
        gemm_bf16_wmma<128, 128, true, 4><<<dim3(8, 32, 1), 256, 0, stream>>>(g);
    }

    // z = LN2(x1) * (1+gamma2) + beta2 -> bf16 (reuse y buffer)
    ln_mod_kernel<<<4096, 256, 0, stream>>>(out, ln2_g, ln2_b, mod + 3 * 4096, mod + 4 * 4096, ybf);

    // h1 = gelu(z @ fc1_w + fc1_b) -> bf16
    {
        GemmP g{};
        g.A = ybf; g.B = Wfc1t; g.bias = fc1_b; g.outB = h1;
        g.K = 1024; g.lda = 1024; g.ldb = 1024; g.ldc = 4096;
        gemm_bf16_wmma<128, 128, true, 3><<<dim3(32, 32, 1), 256, 0, stream>>>(g);
    }

    // out = x1 + (h1 @ fc2_w + fc2_b) * alpha2  (reads+rewrites d_out)
    {
        GemmP g{};
        g.A = h1; g.B = Wfc2t; g.bias = fc2_b; g.resid = out; g.alpha = mod + 5 * 4096;
        g.outF = out; g.K = 4096; g.lda = 4096; g.ldb = 4096; g.ldc = 1024;
        gemm_bf16_wmma<128, 128, true, 4><<<dim3(8, 32, 1), 256, 0, stream>>>(g);
    }
}
